// PharmaGNN_22943715295616
// MI455X (gfx1250) — matrix operand order repaired and checked by reading, not compiled, not run on hardware
//
#include <hip/hip_runtime.h>
#include <hip/hip_bf16.h>

// ---------------------------------------------------------------------------
// PharmaGNN on MI455X (gfx1250). bf16 WMMA GEMMs with LDS-staged A strips and
// pre-transposed B so every fragment load is a 128-bit op; memory-bound GATv2
// edge softmax via native f32/u32 global atomics and float4 traffic;
// degenerate 1-key cross-attention folded to V@Wo; fused LN+leaky+fc2 head.
// ---------------------------------------------------------------------------

typedef __attribute__((ext_vector_type(16))) __bf16 v16bf;
typedef __attribute__((ext_vector_type(8)))  __bf16 v8bf;
typedef __attribute__((ext_vector_type(8)))  float  v8f;

#define DEVINL __device__ __forceinline__

DEVINL float leakyf(float x, float s) { return x > 0.f ? x : s * x; }

// order-preserving float<->uint encoding for atomic max on floats
DEVINL unsigned f2ord(float f) {
  unsigned u = __float_as_uint(f);
  return (u & 0x80000000u) ? ~u : (u | 0x80000000u);
}
DEVINL float ord2f(unsigned u) {
  return __uint_as_float((u & 0x80000000u) ? (u & 0x7FFFFFFFu) : ~u);
}

DEVINL void atomAddF(float* p, float v) {
  __hip_atomic_fetch_add(p, v, __ATOMIC_RELAXED, __HIP_MEMORY_SCOPE_AGENT);
}

// round-to-nearest-even f32 -> bf16 via bit math
DEVINL __bf16 f2bf(float f) {
  unsigned u = __float_as_uint(f);
  unsigned r = (u + 0x7FFFu + ((u >> 16) & 1u)) >> 16;
  unsigned short hs = (unsigned short)r;
  __bf16 b;
  __builtin_memcpy(&b, &hs, 2);
  return b;
}

// ---------------------------------------------------------------------------
__global__ void k_f32_to_bf16(const float* __restrict__ s, __bf16* __restrict__ d, int n) {
  int i = blockIdx.x * blockDim.x + threadIdx.x;
  if (i < n) d[i] = f2bf(s[i]);
}

// Wt[m*K + k] = bf16(W[k*M + m]) : one-shot weight transpose+convert.
__global__ void k_w_transpose(const float* __restrict__ W, __bf16* __restrict__ Wt,
                              int K, int M) {
  int i = blockIdx.x * blockDim.x + threadIdx.x;
  if (i >= K * M) return;
  int m = i / K, k = i - m * K;
  Wt[i] = f2bf(W[(size_t)k * M + m]);
}

// naive linear for tiny K (layer 1: K = 10 or 5)
__global__ void k_linear_small(const float* __restrict__ x, const float* __restrict__ W,
                               float* __restrict__ out, int N, int K, int M) {
  int i = blockIdx.x * blockDim.x + threadIdx.x;
  if (i >= N * M) return;
  int n = i / M, m = i - n * M;
  float acc = 0.f;
  for (int k = 0; k < K; ++k) acc += x[(size_t)n * K + k] * W[(size_t)k * M + m];
  out[i] = acc;
}

// ---------------------------------------------------------------------------
// bf16 WMMA GEMM. C[Rows x Cols] = A[Rows x K] * Bt^T (+bias[col]);
// Bt is pre-transposed [Cols x K] bf16 so B fragments are contiguous.
// Block = 256 threads = 8 waves; grid = (Rows/16, Cols/128). The block's
// 16-row A strip (contiguous 16*K bf16) is staged in LDS with coalesced b128
// copies; every fragment load is then 2x ds_load_b128 (A) or 2x
// global_load_b128 (Bt). K is a template constant -> fully unrolled WMMA chain.
template <int K>
__global__ void k_gemm_bf16_wmma(const __bf16* __restrict__ A, const __bf16* __restrict__ Bt,
                                 float* __restrict__ C, const float* __restrict__ bias,
                                 int Cols) {
  __shared__ __attribute__((aligned(16))) __bf16 As[16 * K];
  const int tid = threadIdx.x;
  const int wave = tid >> 5, lane = tid & 31;
  const int tr = blockIdx.x;
  const int tc = blockIdx.y * 8 + wave;

  // cooperative, fully coalesced copy of the contiguous A strip into LDS
  {
    const uint4* src = (const uint4*)(A + (size_t)tr * 16 * K);
    uint4* dst = (uint4*)As;
    const int chunks = (16 * K) / 8;  // 8 bf16 per uint4
#pragma unroll
    for (int i = tid; i < chunks; i += 256) dst[i] = src[i];
  }
  __syncthreads();

  const int half = lane >> 4;   // K-split selector (ISA 7.12.2 wave32 layout)
  const int l16  = lane & 15;   // A row / B,C column within tile
  const __bf16* Alane = As + l16 * K;
  const __bf16* Brow  = Bt + (size_t)(tc * 16 + l16) * K;

  v8f acc = {};
#pragma unroll
  for (int k0 = 0; k0 < K; k0 += 32) {
    // A 16x32 frag: lanes l / l+16 hold row M=l; K runs {0..7,16..23} / {8..15,24..31}
    v8bf a0 = *(const v8bf*)(Alane + k0 + half * 8);
    v8bf a1 = *(const v8bf*)(Alane + k0 + 16 + half * 8);
    // B 32x16 frag: lane holds column N=l16; K runs {0..7,16..23} / {8..15,24..31}
    v8bf b0 = *(const v8bf*)(Brow + k0 + half * 8);
    v8bf b1 = *(const v8bf*)(Brow + k0 + 16 + half * 8);
    v16bf a = __builtin_shufflevector(a0, a1, 0, 1, 2, 3, 4, 5, 6, 7,
                                      8, 9, 10, 11, 12, 13, 14, 15);
    v16bf b = __builtin_shufflevector(b0, b1, 0, 1, 2, 3, 4, 5, 6, 7,
                                      8, 9, 10, 11, 12, 13, 14, 15);
    acc = __builtin_amdgcn_wmma_f32_16x16x32_bf16(false, a, false, b, (short)0, acc,
                                                  false, false);
  }
  const float bb = bias ? bias[tc * 16 + l16] : 0.f;
  // C/D layout: VGPR r -> row (half*8 + r), column l16
#pragma unroll
  for (int r = 0; r < 8; ++r)
    C[(size_t)(tr * 16 + half * 8 + r) * Cols + tc * 16 + l16] = acc[r] + bb;
}

// ---------------------------------------------------------------------------
// GATv2 edge pipeline. Edge e in [0, E+N): e<E real edge, else self loop e-E.
// Wave per edge; lane covers 8 contiguous channels (2x float4 per row).
__global__ void k_edge_logits(const float* __restrict__ xl, const float* __restrict__ xr,
                              const int* __restrict__ ei, int E, int N, int H,
                              const float* __restrict__ att,
                              float* __restrict__ logits, unsigned* __restrict__ menc) {
  int e = (int)((blockIdx.x * blockDim.x + threadIdx.x) >> 5);
  int lane = threadIdx.x & 31;
  int total = E + N;
  if (e >= total) return;
  int src = (e < E) ? ei[e]     : (e - E);
  int dst = (e < E) ? ei[E + e] : (e - E);
  const float4* pl = (const float4*)(xl + (size_t)src * 256 + lane * 8);
  const float4* pr = (const float4*)(xr + (size_t)dst * 256 + lane * 8);
  const float4* pa = (const float4*)(att + lane * 8);
  float4 l0 = pl[0], l1 = pl[1], r0 = pr[0], r1 = pr[1];
  float4 a0 = pa[0], a1 = pa[1];
  float part = leakyf(l0.x + r0.x, 0.2f) * a0.x + leakyf(l0.y + r0.y, 0.2f) * a0.y +
               leakyf(l0.z + r0.z, 0.2f) * a0.z + leakyf(l0.w + r0.w, 0.2f) * a0.w +
               leakyf(l1.x + r1.x, 0.2f) * a1.x + leakyf(l1.y + r1.y, 0.2f) * a1.y +
               leakyf(l1.z + r1.z, 0.2f) * a1.z + leakyf(l1.w + r1.w, 0.2f) * a1.w;
  part += __shfl_xor(part, 1);
  part += __shfl_xor(part, 2);
  part += __shfl_xor(part, 4);
  part += __shfl_xor(part, 8);
  if (H == 1) part += __shfl_xor(part, 16);
  int group = 32 / H;  // 16 lanes/head (H=2) or 32 (H=1)
  if ((lane & (group - 1)) == 0) {
    int h = lane / group;
    logits[(size_t)e * H + h] = part;
    atomicMax(&menc[(size_t)dst * H + h], f2ord(part));
  }
}

__global__ void k_edge_exp(const float* __restrict__ logits, const unsigned* __restrict__ menc,
                           const int* __restrict__ ei, int E, int N, int H,
                           float* __restrict__ aexp, float* __restrict__ den) {
  int idx = blockIdx.x * blockDim.x + threadIdx.x;
  int total = (E + N) * H;
  if (idx >= total) return;
  int e = idx / H, h = idx - e * H;
  int dst = (e < E) ? ei[E + e] : (e - E);
  float a = __expf(logits[idx] - ord2f(menc[(size_t)dst * H + h]));
  aexp[idx] = a;
  atomAddF(&den[(size_t)dst * H + h], a);
}

__global__ void k_edge_aggr(const float* __restrict__ xl, const float* __restrict__ aexp,
                            const int* __restrict__ ei, int E, int N, int H,
                            float* __restrict__ out) {
  int e = (int)((blockIdx.x * blockDim.x + threadIdx.x) >> 5);
  int lane = threadIdx.x & 31;
  int total = E + N;
  if (e >= total) return;
  int src = (e < E) ? ei[e]     : (e - E);
  int dst = (e < E) ? ei[E + e] : (e - E);
  int h = (lane * 8) / (256 / H);               // head for this lane's channel run
  float w = aexp[(size_t)e * H + h];
  const float4* pl = (const float4*)(xl + (size_t)src * 256 + lane * 8);
  float4 x0 = pl[0], x1 = pl[1];
  float* ob = out + (size_t)dst * 256 + lane * 8;
  atomAddF(ob + 0, x0.x * w); atomAddF(ob + 1, x0.y * w);
  atomAddF(ob + 2, x0.z * w); atomAddF(ob + 3, x0.w * w);
  atomAddF(ob + 4, x1.x * w); atomAddF(ob + 5, x1.y * w);
  atomAddF(ob + 6, x1.z * w); atomAddF(ob + 7, x1.w * w);
}

// out[n,c] = out[n,c]/den[n,h] + bias[c]
__global__ void k_finalize(float* __restrict__ out, const float* __restrict__ den,
                           const float* __restrict__ bias, int N, int H) {
  int i = blockIdx.x * blockDim.x + threadIdx.x;
  if (i >= N * 256) return;
  int n = i >> 8, c = i & 255;
  int Ch = 256 / H;
  out[i] = out[i] / den[(size_t)n * H + c / Ch] + bias[c];
}

// ---------------------------------------------------------------------------
// mean pool over sorted batch ids
__global__ void k_pool_acc(const float* __restrict__ x, const int* __restrict__ batch,
                           float* __restrict__ pooled, float* __restrict__ cnt, int N) {
  int i = blockIdx.x * blockDim.x + threadIdx.x;
  if (i >= N * 256) return;
  int n = i >> 8, c = i & 255;
  int b = batch[n];
  atomAddF(&pooled[(size_t)b * 256 + c], x[i]);
  if (c == 0) atomAddF(&cnt[b], 1.0f);
}

__global__ void k_pool_div(float* __restrict__ pooled, const float* __restrict__ cnt) {
  int i = blockIdx.x * blockDim.x + threadIdx.x;
  if (i >= 256 * 256) return;
  pooled[i] /= fmaxf(cnt[i >> 8], 1.0f);
}

// comb16[b, 0:256]=px, [b, 256:512]=attn (bf16) for the K=512 fc1 WMMA GEMM
__global__ void k_build_comb(const float* __restrict__ px, const float* __restrict__ attn,
                             __bf16* __restrict__ comb) {
  int i = blockIdx.x * blockDim.x + threadIdx.x;
  if (i >= 256 * 512) return;
  int b = i >> 9, c = i & 511;
  comb[i] = f2bf(c < 256 ? px[b * 256 + c] : attn[b * 256 + (c - 256)]);
}

// fused LayerNorm(256) -> LeakyReLU(0.01) -> dot with fc2_w (+fc2_b); wave/row
__global__ void k_ln_head(const float* __restrict__ h, const float* __restrict__ g,
                          const float* __restrict__ be, const float* __restrict__ w2,
                          const float* __restrict__ b2, float* __restrict__ out, int Bg) {
  int row = (int)((blockIdx.x * blockDim.x + threadIdx.x) >> 5);
  int lane = threadIdx.x & 31;
  if (row >= Bg) return;
  float v[8], s = 0.f, sq = 0.f;
#pragma unroll
  for (int j = 0; j < 8; ++j) {
    v[j] = h[(size_t)row * 256 + j * 32 + lane];
    s += v[j];
    sq += v[j] * v[j];
  }
#pragma unroll
  for (int o = 1; o < 32; o <<= 1) { s += __shfl_xor(s, o); sq += __shfl_xor(sq, o); }
  float mean = s * (1.f / 256.f);
  float var  = sq * (1.f / 256.f) - mean * mean;
  float inv  = rsqrtf(var + 1e-5f);
  float acc = 0.f;
#pragma unroll
  for (int j = 0; j < 8; ++j) {
    int c = j * 32 + lane;
    float t = (v[j] - mean) * inv * g[c] + be[c];
    t = leakyf(t, 0.01f);
    acc += t * w2[c];
  }
#pragma unroll
  for (int o = 1; o < 32; o <<= 1) acc += __shfl_xor(acc, o);
  if (lane == 0) out[row] = acc + b2[0];
}

// ---------------------------------------------------------------------------
// Host side
// ---------------------------------------------------------------------------
namespace {

constexpr int NN = 50000;   // nodes per side
constexpr int EE = 400000;  // edges per side
constexpr int DD = 256;
constexpr int BG = 256;

struct Ws {
  float*    XL;   float* XR;   float* X;
  __bf16*   X16;  __bf16* W16a; __bf16* W16b;
  float*    LOG;  float* AEXP;
  unsigned* MENC; float* DEN;
  float*    poolP; float* poolL; float* CNT;
  float*    VBUF;  float* ATTN;
  __bf16*   C16;   float* HBUF;
};

inline Ws carve(void* ws) {
  char* p = (char*)ws;
  auto take = [&](size_t bytes) { char* r = p; p += (bytes + 255) & ~size_t(255); return r; };
  Ws w;
  w.XL    = (float*)take((size_t)NN * DD * 4);
  w.XR    = (float*)take((size_t)NN * DD * 4);
  w.X     = (float*)take((size_t)NN * DD * 4);
  w.X16   = (__bf16*)take((size_t)NN * DD * 2);
  w.W16a  = (__bf16*)take((size_t)512 * 256 * 2);
  w.W16b  = (__bf16*)take((size_t)512 * 256 * 2);
  w.LOG   = (float*)take((size_t)(EE + NN) * 2 * 4);
  w.AEXP  = (float*)take((size_t)(EE + NN) * 2 * 4);
  w.MENC  = (unsigned*)take((size_t)NN * 2 * 4);
  w.DEN   = (float*)take((size_t)NN * 2 * 4);
  w.poolP = (float*)take((size_t)BG * DD * 4);
  w.poolL = (float*)take((size_t)BG * DD * 4);
  w.CNT   = (float*)take((size_t)BG * 4);
  w.VBUF  = (float*)take((size_t)BG * DD * 4);
  w.ATTN  = (float*)take((size_t)BG * DD * 4);
  w.C16   = (__bf16*)take((size_t)BG * 512 * 2);
  w.HBUF  = (float*)take((size_t)BG * DD * 4);
  return w;
}

inline void edge_pipeline(hipStream_t st, const Ws& w, const int* ei,
                          const float* att, const float* bias, int H) {
  const int total = EE + NN;
  const int tND = NN * DD;
  hipMemsetAsync(w.MENC, 0, (size_t)NN * H * 4, st);  // 0 decodes to -NaN: max identity
  hipMemsetAsync(w.DEN,  0, (size_t)NN * H * 4, st);
  hipMemsetAsync(w.X,    0, (size_t)tND * 4, st);
  k_edge_logits<<<(total * 32 + 255) / 256, 256, 0, st>>>(w.XL, w.XR, ei, EE, NN, H,
                                                          att, w.LOG, w.MENC);
  k_edge_exp<<<(total * H + 255) / 256, 256, 0, st>>>(w.LOG, w.MENC, ei, EE, NN, H,
                                                      w.AEXP, w.DEN);
  k_edge_aggr<<<(total * 32 + 255) / 256, 256, 0, st>>>(w.XL, w.AEXP, ei, EE, NN, H, w.X);
  k_finalize<<<(tND + 255) / 256, 256, 0, st>>>(w.X, w.DEN, bias, NN, H);
}

inline void run_side(hipStream_t st, const Ws& w,
                     const float* x0, int K0, const int* ei,
                     const float* W1l, const float* W1r, const float* att1, const float* b1,
                     const float* W2l, const float* W2r, const float* att2, const float* b2) {
  const int tND = NN * DD;
  // layer 1: tiny-K naive transforms, H=2
  k_linear_small<<<(tND + 255) / 256, 256, 0, st>>>(x0, W1l, w.XL, NN, K0, DD);
  k_linear_small<<<(tND + 255) / 256, 256, 0, st>>>(x0, W1r, w.XR, NN, K0, DD);
  edge_pipeline(st, w, ei, att1, b1, 2);
  // layer 2: 256x256 WMMA transforms, H=1  (w.X holds layer-1 output)
  k_f32_to_bf16<<<(tND + 255) / 256, 256, 0, st>>>(w.X, w.X16, tND);
  k_w_transpose<<<(DD * DD + 255) / 256, 256, 0, st>>>(W2l, w.W16a, DD, DD);
  k_w_transpose<<<(DD * DD + 255) / 256, 256, 0, st>>>(W2r, w.W16b, DD, DD);
  dim3 g2(NN / 16, DD / 128);  // 3125 x 2 blocks
  k_gemm_bf16_wmma<256><<<g2, 256, 0, st>>>(w.X16, w.W16a, w.XL, nullptr, DD);
  k_gemm_bf16_wmma<256><<<g2, 256, 0, st>>>(w.X16, w.W16b, w.XR, nullptr, DD);
  edge_pipeline(st, w, ei, att2, b2, 1);
}

inline void mean_pool(hipStream_t st, const Ws& w, const float* x, const int* batch,
                      float* pooled) {
  hipMemsetAsync(pooled, 0, (size_t)BG * DD * 4, st);
  hipMemsetAsync(w.CNT, 0, (size_t)BG * 4, st);
  k_pool_acc<<<(NN * DD + 255) / 256, 256, 0, st>>>(x, batch, pooled, w.CNT, NN);
  k_pool_div<<<(BG * DD + 255) / 256, 256, 0, st>>>(pooled, w.CNT);
}

}  // namespace

extern "C" void kernel_launch(void* const* d_in, const int* in_sizes, int n_in,
                              void* d_out, int out_size, void* d_ws, size_t ws_size,
                              hipStream_t stream) {
  (void)in_sizes; (void)n_in; (void)out_size; (void)ws_size;
  const float* protein_x = (const float*)d_in[0];
  const float* ligand_x  = (const float*)d_in[1];
  const int*   p_ei      = (const int*)d_in[2];
  const int*   l_ei      = (const int*)d_in[3];
  const int*   p_batch   = (const int*)d_in[4];
  const int*   l_batch   = (const int*)d_in[5];
  const float* p1_Wl = (const float*)d_in[6];  const float* p1_Wr = (const float*)d_in[7];
  const float* p1_att= (const float*)d_in[8];  const float* p1_b  = (const float*)d_in[9];
  const float* p2_Wl = (const float*)d_in[10]; const float* p2_Wr = (const float*)d_in[11];
  const float* p2_att= (const float*)d_in[12]; const float* p2_b  = (const float*)d_in[13];
  const float* l1_Wl = (const float*)d_in[14]; const float* l1_Wr = (const float*)d_in[15];
  const float* l1_att= (const float*)d_in[16]; const float* l1_b  = (const float*)d_in[17];
  const float* l2_Wl = (const float*)d_in[18]; const float* l2_Wr = (const float*)d_in[19];
  const float* l2_att= (const float*)d_in[20]; const float* l2_b  = (const float*)d_in[21];
  // d_in[22..25] = Wq,bq,Wk,bk: dead (softmax over a single key is identically 1)
  const float* Wv = (const float*)d_in[26]; const float* bv = (const float*)d_in[27];
  const float* Wo = (const float*)d_in[28]; const float* bo = (const float*)d_in[29];
  const float* fc1_w = (const float*)d_in[30]; const float* fc1_b = (const float*)d_in[31];
  const float* ln_g  = (const float*)d_in[32]; const float* ln_b  = (const float*)d_in[33];
  const float* fc2_w = (const float*)d_in[34]; const float* fc2_b = (const float*)d_in[35];

  Ws w = carve(d_ws);

  // protein tower, then pool (buffers are recycled between sides)
  run_side(stream, w, protein_x, 10, p_ei, p1_Wl, p1_Wr, p1_att, p1_b,
           p2_Wl, p2_Wr, p2_att, p2_b);
  mean_pool(stream, w, w.X, p_batch, w.poolP);

  // ligand tower, then pool
  run_side(stream, w, ligand_x, 5, l_ei, l1_Wl, l1_Wr, l1_att, l1_b,
           l2_Wl, l2_Wr, l2_att, l2_b);
  mean_pool(stream, w, w.X, l_batch, w.poolL);

  // cross attention with a single key: softmax == 1, so attn = (lx@Wv+bv)@Wo+bo
  const int sq = DD * DD;  // 65536
  __bf16* LX16 = w.X16;    // reuse big bf16 buffer
  __bf16* V16  = w.X16 + sq;
  dim3 gAttn(BG / 16, DD / 128);  // 16 x 2 blocks
  k_f32_to_bf16<<<(sq + 255) / 256, 256, 0, stream>>>(w.poolL, LX16, sq);
  k_w_transpose<<<(sq + 255) / 256, 256, 0, stream>>>(Wv, w.W16a, DD, DD);
  k_gemm_bf16_wmma<256><<<gAttn, 256, 0, stream>>>(LX16, w.W16a, w.VBUF, bv, DD);
  k_f32_to_bf16<<<(sq + 255) / 256, 256, 0, stream>>>(w.VBUF, V16, sq);
  k_w_transpose<<<(sq + 255) / 256, 256, 0, stream>>>(Wo, w.W16a, DD, DD);
  k_gemm_bf16_wmma<256><<<gAttn, 256, 0, stream>>>(V16, w.W16a, w.ATTN, bo, DD);

  // comb = [px, attn] (bf16), fc1 (K=512) via WMMA, then fused LN+leaky+fc2
  k_build_comb<<<(BG * 512 + 255) / 256, 256, 0, stream>>>(w.poolP, w.ATTN, w.C16);
  k_w_transpose<<<(512 * 256 + 255) / 256, 256, 0, stream>>>(fc1_w, w.W16a, 512, 256);
  k_gemm_bf16_wmma<512><<<gAttn, 256, 0, stream>>>(w.C16, w.W16a, w.HBUF, fc1_b, DD);
  k_ln_head<<<(BG * 32 + 255) / 256, 256, 0, stream>>>(w.HBUF, ln_g, ln_b, fc2_w, fc2_b,
                                                       (float*)d_out, BG);
}
